// StableMultiHeadAttention_16037407883677
// MI455X (gfx1250) — matrix operand order repaired and checked by reading, not compiled, not run on hardware
//
#include <hip/hip_runtime.h>

typedef __bf16 bf16_t;
typedef __attribute__((ext_vector_type(16))) __bf16 bf16x16;
typedef __attribute__((ext_vector_type(8)))  __bf16 bf16x8;
typedef __attribute__((ext_vector_type(4)))  __bf16 bf16x4;
typedef __attribute__((ext_vector_type(8)))  float  f32x8;

#define WMMA_BF16(a, b, c) \
  __builtin_amdgcn_wmma_f32_16x16x32_bf16(false, (a), false, (b), (short)0, (c), false, false)

#define CAT16(lo, hi) __builtin_shufflevector((lo), (hi), 0,1,2,3,4,5,6,7,8,9,10,11,12,13,14,15)

// ---------------------------------------------------------------------------
// f32 -> bf16 elementwise conversion (vectorized x4)
// ---------------------------------------------------------------------------
__global__ void cvt_f32_to_bf16(const float* __restrict__ src,
                                bf16_t* __restrict__ dst, int n) {
  int i = (blockIdx.x * blockDim.x + threadIdx.x) * 4;
  if (i >= n) return;
  float4 v = *(const float4*)(src + i);
  bf16x4 o;
  o[0] = (bf16_t)v.x; o[1] = (bf16_t)v.y; o[2] = (bf16_t)v.z; o[3] = (bf16_t)v.w;
  *(bf16x4*)(dst + i) = o;
}

// ---------------------------------------------------------------------------
// QKV projection: qkv[4096,3072] = Xbf[4096,1024] @ Wqkv^T  (W row-major [3072,1024])
// Each wave: 32(M) x 64(N) tile -> 2x4 WMMA accumulators, K step 32.
// N-tile of 64 maps exactly to one (which, head); epilogue scatters:
//   which==0 -> Q (pre-scaled by 1/sqrt(Dh)), row-major [S,64] per (b,h)
//   which==1 -> K, row-major [S,64] per (b,h)
//   which==2 -> V^T, [64,S] per (b,h)  (vectorized 16B stores along S)
// ---------------------------------------------------------------------------
__global__ __launch_bounds__(256, 1)
void qkv_gemm(const bf16_t* __restrict__ X, const bf16_t* __restrict__ W,
              bf16_t* __restrict__ Qd, bf16_t* __restrict__ Kd,
              bf16_t* __restrict__ VTd) {
  const int lane = threadIdx.x & 31, wave = threadIdx.x >> 5;
  const int l16 = lane & 15, hl = lane >> 4;
  const int gw = blockIdx.x * 8 + wave;
  const int NT = 48;                       // 3072 / 64
  const int Mbase = (gw / NT) * 32;
  const int Nbase = (gw % NT) * 64;

  f32x8 acc[2][4] = {};
  for (int kb = 0; kb < 1024; kb += 32) {
    bf16x16 a[2], b[4];
#pragma unroll
    for (int t = 0; t < 2; ++t) {
      const bf16_t* ap = X + (size_t)(Mbase + t * 16 + l16) * 1024 + kb + hl * 16;
      bf16x8 lo = *(const bf16x8*)ap;
      bf16x8 hi = *(const bf16x8*)(ap + 8);
      a[t] = CAT16(lo, hi);
    }
#pragma unroll
    for (int j = 0; j < 4; ++j)
      b[j] = *(const bf16x16*)(W + (size_t)(Nbase + j * 16 + l16) * 1024 + kb + hl * 16);
#pragma unroll
    for (int t = 0; t < 2; ++t)
#pragma unroll
      for (int j = 0; j < 4; ++j)
        acc[t][j] = WMMA_BF16(a[t], b[j], acc[t][j]);
  }

  const int which = Nbase >> 10;           // 0=q 1=k 2=v
  const int h     = (Nbase & 1023) >> 6;
  const int bIdx  = Mbase >> 11;           // token / 2048
  const int sBase = Mbase & 2047;
  const size_t bhQ = ((size_t)bIdx * 16 + h) * 2048 * 64;

  if (which == 2) {
    bf16_t* vt = VTd + ((size_t)bIdx * 16 + h) * (size_t)64 * 2048;
#pragma unroll
    for (int t = 0; t < 2; ++t)
#pragma unroll
      for (int j = 0; j < 4; ++j) {
        int d = j * 16 + l16;
        bf16x8 pk;
#pragma unroll
        for (int r = 0; r < 8; ++r) pk[r] = (bf16_t)acc[t][j][r];
        *(bf16x8*)(vt + (size_t)d * 2048 + sBase + t * 16 + hl * 8) = pk;
      }
  } else {
    const float sc = (which == 0) ? 0.125f : 1.0f;   // 1/sqrt(64) folded into Q
    bf16_t* dst = (which == 0 ? Qd : Kd) + bhQ;
#pragma unroll
    for (int t = 0; t < 2; ++t)
#pragma unroll
      for (int j = 0; j < 4; ++j) {
        int d = j * 16 + l16;
#pragma unroll
        for (int r = 0; r < 8; ++r)
          dst[(size_t)(sBase + t * 16 + hl * 8 + r) * 64 + d] =
              (bf16_t)(acc[t][j][r] * sc);
      }
  }
}

// ---------------------------------------------------------------------------
// Flash-style attention. One wave = 16 queries of one (b,h); block = 128 queries.
// KV stepped by 32 keys: 4 WMMA (scores) + online softmax + 4 WMMA (P @ V).
// P is converted from C-layout to A-layout via a 1KB/wave LDS round-trip.
// Softmax matches reference: running max (== global max), clamp >= -50,
// denominator sum + 1e-6.
// ---------------------------------------------------------------------------
__global__ __launch_bounds__(256, 1)
void attn_fwd(const bf16_t* __restrict__ Qg, const bf16_t* __restrict__ Kg,
              const bf16_t* __restrict__ VTg, bf16_t* __restrict__ Og) {
  __shared__ bf16_t plds[8][16 * 32];
  const int lane = threadIdx.x & 31, wave = threadIdx.x >> 5;
  const int l16 = lane & 15, hl = lane >> 4;
  const int bh = blockIdx.x >> 4;          // 16 q-blocks per (b,h)
  const int qblk = blockIdx.x & 15;
  const int Qbase = qblk * 128 + wave * 16;

  const bf16_t* q  = Qg  + (size_t)bh * 2048 * 64;
  const bf16_t* k  = Kg  + (size_t)bh * 2048 * 64;
  const bf16_t* vt = VTg + (size_t)bh * 64 * 2048;

  bf16x16 aq[2];
#pragma unroll
  for (int f = 0; f < 2; ++f) {
    const bf16_t* ap = q + (size_t)(Qbase + l16) * 64 + f * 32 + hl * 16;
    bf16x8 lo = *(const bf16x8*)ap;
    bf16x8 hi = *(const bf16x8*)(ap + 8);
    aq[f] = CAT16(lo, hi);
  }

  float mrow[8], ssum[8];
#pragma unroll
  for (int r = 0; r < 8; ++r) { mrow[r] = -3.0e38f; ssum[r] = 0.0f; }
  f32x8 o[4] = {};

  for (int kb = 0; kb < 2048; kb += 32) {
    if (kb + 32 < 2048) {
      __builtin_prefetch(k + (size_t)(kb + 32 + l16) * 64, 0, 0);
      __builtin_prefetch(vt + (size_t)l16 * 2048 + kb + 32, 0, 0);
    }
    // K block as B fragments: keys (cols) = kb..kb+31, dims (k-dim) split 0-31/32-63
    bf16x16 bk00 = *(const bf16x16*)(k + (size_t)(kb + l16) * 64 + hl * 16);
    bf16x16 bk01 = *(const bf16x16*)(k + (size_t)(kb + l16) * 64 + 32 + hl * 16);
    bf16x16 bk10 = *(const bf16x16*)(k + (size_t)(kb + 16 + l16) * 64 + hl * 16);
    bf16x16 bk11 = *(const bf16x16*)(k + (size_t)(kb + 16 + l16) * 64 + 32 + hl * 16);

    f32x8 s0 = {}, s1 = {};
    s0 = WMMA_BF16(aq[0], bk00, s0);
    s0 = WMMA_BF16(aq[1], bk01, s0);
    s1 = WMMA_BF16(aq[0], bk10, s1);
    s1 = WMMA_BF16(aq[1], bk11, s1);

    float resc[8];
#pragma unroll
    for (int r = 0; r < 8; ++r) {
      // row r+8*hl: reduce across the 16 lanes of this half (xor<16 stays in half)
      float cur = fmaxf(s0[r], s1[r]);
      cur = fmaxf(cur, __shfl_xor(cur, 1));
      cur = fmaxf(cur, __shfl_xor(cur, 2));
      cur = fmaxf(cur, __shfl_xor(cur, 4));
      cur = fmaxf(cur, __shfl_xor(cur, 8));
      float mnew = fmaxf(mrow[r], cur);
      float rs = __expf(mrow[r] - mnew);           // 0 on first step
      float p0 = __expf(fmaxf(s0[r] - mnew, -50.f));
      float p1 = __expf(fmaxf(s1[r] - mnew, -50.f));
      ssum[r] = ssum[r] * rs + p0 + p1;            // lane-partial row sum
      resc[r] = rs;
      mrow[r] = mnew;
      int row = r + 8 * hl;
      plds[wave][row * 32 + l16]      = (bf16_t)p0;
      plds[wave][row * 32 + 16 + l16] = (bf16_t)p1;
    }
#pragma unroll
    for (int j = 0; j < 4; ++j)
#pragma unroll
      for (int r = 0; r < 8; ++r) o[j][r] *= resc[r];

    // wave-local LDS RAW: wait for the P stores before A-layout reload
    asm volatile("s_wait_dscnt 0" ::: "memory");
    bf16x8 plo = *(const bf16x8*)&plds[wave][l16 * 32 + hl * 16];
    bf16x8 phi = *(const bf16x8*)&plds[wave][l16 * 32 + hl * 16 + 8];
    bf16x16 pa = CAT16(plo, phi);

#pragma unroll
    for (int j = 0; j < 4; ++j) {
      bf16x16 bv = *(const bf16x16*)(vt + (size_t)(j * 16 + l16) * 2048 + kb + hl * 16);
      o[j] = WMMA_BF16(pa, bv, o[j]);
    }
  }

#pragma unroll
  for (int r = 0; r < 8; ++r) {
    float s = ssum[r];
    s += __shfl_xor(s, 1);
    s += __shfl_xor(s, 2);
    s += __shfl_xor(s, 4);
    s += __shfl_xor(s, 8);
    ssum[r] = 1.0f / (s + 1e-6f);                  // reference: sum + EPS
  }

  const int hIdx = bh & 15, bIdx = bh >> 4;
  bf16_t* dst = Og + (size_t)bIdx * 2048 * 1024 + hIdx * 64;
#pragma unroll
  for (int j = 0; j < 4; ++j)
#pragma unroll
    for (int r = 0; r < 8; ++r) {
      int srow = Qbase + hl * 8 + r;
      dst[(size_t)srow * 1024 + j * 16 + l16] = (bf16_t)(o[j][r] * ssum[r]);
    }
}

// ---------------------------------------------------------------------------
// Output projection: out[4096,1024] = attn_bf[4096,1024] @ Wout^T, fp32 out.
// Same 32x64-per-wave tiling as qkv_gemm.
// ---------------------------------------------------------------------------
__global__ __launch_bounds__(256, 1)
void out_gemm(const bf16_t* __restrict__ A, const bf16_t* __restrict__ W,
              float* __restrict__ Cout) {
  const int lane = threadIdx.x & 31, wave = threadIdx.x >> 5;
  const int l16 = lane & 15, hl = lane >> 4;
  const int gw = blockIdx.x * 8 + wave;
  const int NT = 16;                       // 1024 / 64
  const int Mbase = (gw / NT) * 32;
  const int Nbase = (gw % NT) * 64;

  f32x8 acc[2][4] = {};
  for (int kb = 0; kb < 1024; kb += 32) {
    bf16x16 a[2], b[4];
#pragma unroll
    for (int t = 0; t < 2; ++t) {
      const bf16_t* ap = A + (size_t)(Mbase + t * 16 + l16) * 1024 + kb + hl * 16;
      bf16x8 lo = *(const bf16x8*)ap;
      bf16x8 hi = *(const bf16x8*)(ap + 8);
      a[t] = CAT16(lo, hi);
    }
#pragma unroll
    for (int j = 0; j < 4; ++j)
      b[j] = *(const bf16x16*)(W + (size_t)(Nbase + j * 16 + l16) * 1024 + kb + hl * 16);
#pragma unroll
    for (int t = 0; t < 2; ++t)
#pragma unroll
      for (int j = 0; j < 4; ++j)
        acc[t][j] = WMMA_BF16(a[t], b[j], acc[t][j]);
  }
#pragma unroll
  for (int t = 0; t < 2; ++t)
#pragma unroll
    for (int j = 0; j < 4; ++j)
#pragma unroll
      for (int r = 0; r < 8; ++r)
        Cout[(size_t)(Mbase + t * 16 + hl * 8 + r) * 1024 + Nbase + j * 16 + l16] =
            acc[t][j][r];
}

// ---------------------------------------------------------------------------
extern "C" void kernel_launch(void* const* d_in, const int* in_sizes, int n_in,
                              void* d_out, int out_size, void* d_ws, size_t ws_size,
                              hipStream_t stream) {
  const float* x    = (const float*)d_in[0];   // [2,2048,1024]
  const float* wqkv = (const float*)d_in[1];   // [3072,1024]
  const float* wout = (const float*)d_in[2];   // [1024,1024]
  float* out = (float*)d_out;

  char* ws = (char*)d_ws;
  size_t off = 0;
  auto carve = [&](size_t bytes) {
    char* p = ws + off;
    off += (bytes + 255) & ~(size_t)255;
    return p;
  };
  bf16_t* x_bf    = (bf16_t*)carve(4096ull * 1024 * 2);
  bf16_t* wqkv_bf = (bf16_t*)carve(3072ull * 1024 * 2);
  bf16_t* wout_bf = (bf16_t*)carve(1024ull * 1024 * 2);
  bf16_t* q_bf    = (bf16_t*)carve(32ull * 2048 * 64 * 2);
  bf16_t* k_bf    = (bf16_t*)carve(32ull * 2048 * 64 * 2);
  bf16_t* vt_bf   = (bf16_t*)carve(32ull * 2048 * 64 * 2);
  bf16_t* ao_bf   = (bf16_t*)carve(4096ull * 1024 * 2);

  cvt_f32_to_bf16<<<4096, 256, 0, stream>>>(x, x_bf, 4096 * 1024);
  cvt_f32_to_bf16<<<3072, 256, 0, stream>>>(wqkv, wqkv_bf, 3072 * 1024);
  cvt_f32_to_bf16<<<1024, 256, 0, stream>>>(wout, wout_bf, 1024 * 1024);

  qkv_gemm<<<768, 256, 0, stream>>>(x_bf, wqkv_bf, q_bf, k_bf, vt_bf);
  attn_fwd<<<512, 256, 0, stream>>>(q_bf, k_bf, vt_bf, ao_bf);
  out_gemm<<<256, 256, 0, stream>>>(ao_bf, wout_bf, out);
}
